// QuantumLLM_43044162240984
// MI455X (gfx1250) — matrix operand order repaired and checked by reading, not compiled, not run on hardware
//
#include <hip/hip_runtime.h>
#include <hip/hip_bf16.h>
#include <math.h>

#define DIM 2048
#define H 16
#define HD 128
#define LAYERS 6
#define VOCAB 105
#define NPAD 112            // 105 padded to 7 tiles of 16
#define NPOS (8 * 2048)
#define PI_F 3.14159265358979323846f
#define TWO_PI_F 6.283185307179586f

typedef __attribute__((ext_vector_type(16))) _Float16 v16h;
typedef __attribute__((ext_vector_type(8)))  float    v8f;
typedef __attribute__((ext_vector_type(4)))  _Float16 h4;

__device__ __forceinline__ float wave_red(float v) {
  #pragma unroll
  for (int o = 16; o > 0; o >>= 1) v += __shfl_xor(v, o, 32);
  return v;
}

// ---------------- init kernels (tiny, run once per launch) ----------------

// patterns[h][d] = sin(atan2(xn[d], xn[(d-1)%128])), xn = (t+phase)/||.||
__global__ void k_init_patterns(float* patt) {
  int h = blockIdx.x;       // 16 blocks
  int d = threadIdx.x;      // 128 threads
  float ph = TWO_PI_F * (float)h / 16.f;
  float xd = TWO_PI_F * (float)d / 127.f + ph;
  __shared__ float red[4];
  float s = wave_red(xd * xd);
  int lane = threadIdx.x & 31, wid = threadIdx.x >> 5;
  if (lane == 0) red[wid] = s;
  __syncthreads();
  float nrm = sqrtf(red[0] + red[1] + red[2] + red[3]) + 1e-8f;
  int dp = (d + 127) & 127;
  float xp = TWO_PI_F * (float)dp / 127.f + ph;
  patt[h * HD + d] = sinf(atan2f(xd / nrm, xp / nrm));
}

__global__ void k_init_hv(float* hp, float* vp) {
  int i = blockIdx.x * blockDim.x + threadIdx.x;
  if (i < DIM) {
    float t = PI_F * (float)i / (float)(DIM - 1);
    hp[i] = sinf(t);
    vp[i] = cosf(t);
  }
}

__global__ void k_init_wpad(const float* w, _Float16* wpad) {
  int i = blockIdx.x * blockDim.x + threadIdx.x;  // over DIM*NPAD
  if (i < DIM * NPAD) {
    int k = i / NPAD, n = i % NPAD;
    wpad[i] = (n < VOCAB) ? (_Float16)w[k * VOCAB + n] : (_Float16)0.f;
  }
}

// ---------------- embed + input LayerNorm ----------------

__global__ void __launch_bounds__(128) k_embed_ln(
    const int* __restrict__ tokens, const float* __restrict__ emb,
    const float* __restrict__ pos, const float* __restrict__ g,
    const float* __restrict__ b, float* __restrict__ xg) {
  __shared__ __align__(16) float s_x[4][DIM];
  int wid = threadIdx.x >> 5, lane = threadIdx.x & 31;
  int p = blockIdx.x * 4 + wid;            // one wave per position
  int l = p & 2047;
  int tok = tokens[p];
  const float4* er = (const float4*)(emb + (size_t)tok * DIM);
  const float4* pr = (const float4*)(pos + (size_t)l * DIM);
  float4* sx4 = (float4*)&s_x[wid][0];
  float sum = 0.f, sq = 0.f;
  for (int c = 0; c < 16; ++c) {           // 128-bit per-lane row traffic
    int i = lane + 32 * c;
    float4 e = er[i], q = pr[i];
    float4 v = make_float4(e.x + q.x, e.y + q.y, e.z + q.z, e.w + q.w);
    sx4[i] = v;
    sum += v.x + v.y + v.z + v.w;
    sq += v.x * v.x + v.y * v.y + v.z * v.z + v.w * v.w;
  }
  sum = wave_red(sum); sq = wave_red(sq);
  float mean = sum / (float)DIM;
  float rstd = rsqrtf(sq / (float)DIM - mean * mean + 1e-5f);
  float4* xr4 = (float4*)(xg + (size_t)p * DIM);
  const float4* g4 = (const float4*)g;
  const float4* b4 = (const float4*)b;
  for (int c = 0; c < 16; ++c) {
    int i = lane + 32 * c;
    float4 v = sx4[i], gg = g4[i], bb = b4[i];
    xr4[i] = make_float4((v.x - mean) * rstd * gg.x + bb.x,
                         (v.y - mean) * rstd * gg.y + bb.y,
                         (v.z - mean) * rstd * gg.z + bb.z,
                         (v.w - mean) * rstd * gg.w + bb.w);
  }
}

// ---------------- fused layer: attn + residual + LN + encode_state (in place) ----------------

__global__ void __launch_bounds__(128) k_layer(
    float* __restrict__ xg, const float* __restrict__ patt,
    const float* __restrict__ ln_g, const float* __restrict__ ln_b,
    const float* __restrict__ hp, const float* __restrict__ vp) {
  __shared__ __align__(16) float    s_x[4][DIM];      // f32 row (residual / LN workspace)
  __shared__ __align__(16) _Float16 s_h[4][DIM];      // phase-modulated X4, f16
  __shared__ __align__(16) float    s_sc[4][H * H];   // 16x16 scores / probs
  int wid = threadIdx.x >> 5, lane = threadIdx.x & 31;
  int hi  = (lane >> 4) & 1;            // half-wave select (ISA A/B/D layouts)
  int l15 = lane & 15;
  int p = blockIdx.x * 4 + wid;         // one wave per token position
  float* xr = xg + (size_t)p * DIM;
  float4* xr4 = (float4*)xr;
  float4* sx4 = (float4*)&s_x[wid][0];
  h4* sh4 = (h4*)&s_h[wid][0];
  const float4* patt4 = (const float4*)patt;

  // Stage the f32 row global -> LDS with the CDNA5 async data mover
  // (GLOBAL_LOAD_ASYNC_TO_LDS_B128, ASYNCcnt-tracked): no VGPR round trip.
  {
    // low 32 bits of a generic pointer to LDS == LDS byte offset
    unsigned int lbase = (unsigned int)(unsigned long long)&s_x[wid][0];
    #pragma unroll
    for (int c = 0; c < 16; ++c) {
      int i = lane + 32 * c;
      unsigned int loff = lbase + (unsigned int)i * 16u;
      unsigned long long ga = (unsigned long long)(xr4 + i);
      asm volatile("global_load_async_to_lds_b128 %0, %1, off"
                   :: "v"(loff), "v"(ga) : "memory");
    }
    asm volatile("s_wait_asynccnt 0" ::: "memory");
  }
  __syncthreads();

  // build X4 = x * patterns (f16) from the staged row
  for (int c = 0; c < 16; ++c) {
    int i = lane + 32 * c;
    float4 v = sx4[i];
    float4 pp = patt4[i];
    h4 hh;
    hh[0] = (_Float16)(v.x * pp.x); hh[1] = (_Float16)(v.y * pp.y);
    hh[2] = (_Float16)(v.z * pp.z); hh[3] = (_Float16)(v.w * pp.w);
    sh4[i] = hh;
  }
  __syncthreads();

  // scores = X4 @ X4^T / sqrt(128): 4 chained v_wmma_f32_16x16x32_f16 (K=128)
  v8f acc = {0, 0, 0, 0, 0, 0, 0, 0};
  #pragma unroll
  for (int c = 0; c < 4; ++c) {
    v16h a, b;
    int arow = l15 * HD + c * 32;
    #pragma unroll
    for (int j = 0; j < 16; ++j) {
      // A 16x32 f16, this code's contiguous k order: lanes 0-15 K=0-15, lanes 16-31 K=16-31
      a[j] = s_h[wid][arow + hi * 16 + j];
      // B 32x16 layout: lane -> column n (=head g), halves -> K, +16 for hi half
      b[j] = s_h[wid][arow + hi * 16 + j];
    }
    acc = __builtin_amdgcn_wmma_f32_16x16x32_f16(false, a, false, b, (short)0, acc, false, false);
  }
  const float scale = 0.08838834764831845f;  // 1/sqrt(128)
  #pragma unroll
  for (int r = 0; r < 8; ++r)                // D layout: lane=N, vgpr=M (M+8 for hi)
    s_sc[wid][(hi * 8 + r) * 16 + l15] = acc[r] * scale;
  __syncthreads();

  // row softmax over g (16 rows handled by lanes 0-15)
  if (lane < 16) {
    float* row = &s_sc[wid][lane * 16];
    float mx = row[0];
    for (int gi = 1; gi < 16; ++gi) mx = fmaxf(mx, row[gi]);
    float s = 0.f;
    for (int gi = 0; gi < 16; ++gi) { float e = __expf(row[gi] - mx); row[gi] = e; s += e; }
    float inv = 1.f / s;
    for (int gi = 0; gi < 16; ++gi) row[gi] *= inv;
  }
  __syncthreads();

  // out = probs @ X4 : 8 N-tiles, K=16 zero-padded to 32; residual add into s_x
  v16h a2;
  #pragma unroll
  for (int j = 0; j < 16; ++j)  // K>=16 of probs is zero padding
    a2[j] = (hi == 0) ? (_Float16)s_sc[wid][l15 * 16 + j] : (_Float16)0.f;
  #pragma unroll
  for (int t = 0; t < 8; ++t) {
    v16h b2;
    #pragma unroll
    for (int j = 0; j < 16; ++j)  // hi half holds K=16..31 -> zero
      b2[j] = (lane < 16) ? s_h[wid][j * HD + t * 16 + l15] : (_Float16)0.f;
    v8f d = {0, 0, 0, 0, 0, 0, 0, 0};
    d = __builtin_amdgcn_wmma_f32_16x16x32_f16(false, a2, false, b2, (short)0, d, false, false);
    #pragma unroll
    for (int r = 0; r < 8; ++r)
      s_x[wid][(hi * 8 + r) * HD + t * 16 + l15] += d[r];
  }
  __syncthreads();

  // LayerNorm of (x + attn)
  float sum = 0.f, sq = 0.f;
  for (int c = 0; c < 16; ++c) {
    int i = lane + 32 * c;
    float4 v = sx4[i];
    sum += v.x + v.y + v.z + v.w;
    sq += v.x * v.x + v.y * v.y + v.z * v.z + v.w * v.w;
  }
  sum = wave_red(sum); sq = wave_red(sq);
  float mean = sum / (float)DIM;
  float rstd = rsqrtf(sq / (float)DIM - mean * mean + 1e-5f);

  // encode_state: proj = 0.5*(y.hp + y.vp); out = proj + y; L2 normalize
  const float4* hp4 = (const float4*)hp;
  const float4* vp4 = (const float4*)vp;
  const float4* g4 = (const float4*)ln_g;
  const float4* b4 = (const float4*)ln_b;
  float pj = 0.f;
  for (int c = 0; c < 16; ++c) {
    int i = lane + 32 * c;
    float4 v = sx4[i], gg = g4[i], bb = b4[i], hh = hp4[i], vv = vp4[i];
    float4 y = make_float4((v.x - mean) * rstd * gg.x + bb.x,
                           (v.y - mean) * rstd * gg.y + bb.y,
                           (v.z - mean) * rstd * gg.z + bb.z,
                           (v.w - mean) * rstd * gg.w + bb.w);
    sx4[i] = y;
    pj += y.x * (hh.x + vv.x) + y.y * (hh.y + vv.y) +
          y.z * (hh.z + vv.z) + y.w * (hh.w + vv.w);
  }
  pj = 0.5f * wave_red(pj);
  float nq = 0.f;
  for (int c = 0; c < 16; ++c) {
    int i = lane + 32 * c;
    float4 y = sx4[i];
    float4 o = make_float4(y.x + pj, y.y + pj, y.z + pj, y.w + pj);
    sx4[i] = o;
    nq += o.x * o.x + o.y * o.y + o.z * o.z + o.w * o.w;
  }
  float inv = 1.f / (sqrtf(wave_red(nq)) + 1e-8f);
  for (int c = 0; c < 16; ++c) {
    int i = lane + 32 * c;
    float4 o = sx4[i];
    xr4[i] = make_float4(o.x * inv, o.y * inv, o.z * inv, o.w * inv);  // in-place b128 store
  }
}

// ---------------- final LN + projection to VOCAB via WMMA ----------------

__global__ void __launch_bounds__(128) k_final(
    const float* __restrict__ xg, const float* __restrict__ g,
    const float* __restrict__ bb, const _Float16* __restrict__ wpad,
    const float* __restrict__ b_out, float* __restrict__ out) {
  __shared__ float s_mean[16], s_rstd[16];
  __shared__ __align__(16) _Float16 s_a[16 * 32];   // normalized 16x32 K-chunk, f16
  int wid = threadIdx.x >> 5, lane = threadIdx.x & 31;
  int hi = (lane >> 4) & 1, l15 = lane & 15;
  int p0 = blockIdx.x * 16;           // 16-row M tile per block

  // per-row LN stats (each wave: 4 rows), b128 reads
  for (int q = 0; q < 4; ++q) {
    int r = wid * 4 + q;
    const float4* xr4 = (const float4*)(xg + (size_t)(p0 + r) * DIM);
    float sum = 0.f, sq = 0.f;
    for (int c = 0; c < 16; ++c) {
      float4 v = xr4[lane + 32 * c];
      sum += v.x + v.y + v.z + v.w;
      sq += v.x * v.x + v.y * v.y + v.z * v.z + v.w * v.w;
    }
    sum = wave_red(sum); sq = wave_red(sq);
    if (lane == 0) {
      float mean = sum / (float)DIM;
      s_mean[r] = mean;
      s_rstd[r] = rsqrtf(sq / (float)DIM - mean * mean + 1e-5f);
    }
  }
  __syncthreads();

  // N=112 (7 tiles of 16): wave w owns tiles {w, w+4}; wave 3 owns {3}
  int t0 = wid;
  int nt = (wid < 3) ? 2 : 1;
  v8f acc0 = {0, 0, 0, 0, 0, 0, 0, 0}, acc1 = {0, 0, 0, 0, 0, 0, 0, 0};

  const float4* g4 = (const float4*)g;
  const float4* bb4 = (const float4*)bb;
  for (int kc = 0; kc < 64; ++kc) {
    int kb = kc * 32;
    // cooperative: load + normalize 16x32 chunk of x into LDS (f16), b128 loads
    {
      int row = threadIdx.x >> 3;            // 0..15
      int colb = (threadIdx.x & 7) * 4;      // 0,4,..,28
      int k = kb + colb;
      float4 v = *(const float4*)(xg + (size_t)(p0 + row) * DIM + k);
      float4 gg = g4[k >> 2], bc = bb4[k >> 2];
      float mean = s_mean[row], rstd = s_rstd[row];
      h4 hh;
      hh[0] = (_Float16)((v.x - mean) * rstd * gg.x + bc.x);
      hh[1] = (_Float16)((v.y - mean) * rstd * gg.y + bc.y);
      hh[2] = (_Float16)((v.z - mean) * rstd * gg.z + bc.z);
      hh[3] = (_Float16)((v.w - mean) * rstd * gg.w + bc.w);
      *(h4*)&s_a[row * 32 + colb] = hh;
    }
    __syncthreads();
    v16h a;
    #pragma unroll
    for (int j = 0; j < 16; ++j)
      a[j] = s_a[l15 * 32 + hi * 16 + j];
    {
      v16h b;
      #pragma unroll
      for (int j = 0; j < 16; ++j)
        b[j] = wpad[(size_t)(kb + hi * 16 + j) * NPAD + t0 * 16 + l15];
      acc0 = __builtin_amdgcn_wmma_f32_16x16x32_f16(false, a, false, b, (short)0, acc0, false, false);
    }
    if (nt == 2) {  // wave-uniform branch, EXEC stays all-ones inside
      v16h b;
      #pragma unroll
      for (int j = 0; j < 16; ++j)
        b[j] = wpad[(size_t)(kb + hi * 16 + j) * NPAD + (t0 + 4) * 16 + l15];
      acc1 = __builtin_amdgcn_wmma_f32_16x16x32_f16(false, a, false, b, (short)0, acc1, false, false);
    }
    __syncthreads();
  }

  #pragma unroll
  for (int r = 0; r < 8; ++r) {
    int m = hi * 8 + r;
    int n = t0 * 16 + l15;
    if (n < VOCAB) out[(size_t)(p0 + m) * VOCAB + n] = acc0[r] + b_out[n];
  }
  if (nt == 2) {
    #pragma unroll
    for (int r = 0; r < 8; ++r) {
      int m = hi * 8 + r;
      int n = (t0 + 4) * 16 + l15;
      if (n < VOCAB) out[(size_t)(p0 + m) * VOCAB + n] = acc1[r] + b_out[n];
    }
  }
}

// ---------------- launch ----------------

extern "C" void kernel_launch(void* const* d_in, const int* in_sizes, int n_in,
                              void* d_out, int out_size, void* d_ws, size_t ws_size,
                              hipStream_t stream) {
  const int*   tokens = (const int*)  d_in[0];
  const float* emb    = (const float*)d_in[1];
  const float* pos    = (const float*)d_in[2];
  const float* in_g   = (const float*)d_in[3];
  const float* in_b   = (const float*)d_in[4];
  const float* ln_g   = (const float*)d_in[5];
  const float* ln_b   = (const float*)d_in[6];
  const float* out_g  = (const float*)d_in[7];
  const float* out_b  = (const float*)d_in[8];
  const float* w_out  = (const float*)d_in[9];
  const float* b_out  = (const float*)d_in[10];
  float* out = (float*)d_out;

  // workspace layout: x (128 MiB) | patterns | hp | vp | wpad(f16)
  char* ws = (char*)d_ws;
  float* xbuf = (float*)ws;            ws += (size_t)NPOS * DIM * sizeof(float);
  float* patt = (float*)ws;            ws += DIM * sizeof(float);
  float* hp   = (float*)ws;            ws += DIM * sizeof(float);
  float* vp   = (float*)ws;            ws += DIM * sizeof(float);
  _Float16* wpad = (_Float16*)ws;

  k_init_patterns<<<H, HD, 0, stream>>>(patt);
  k_init_hv<<<(DIM + 255) / 256, 256, 0, stream>>>(hp, vp);
  k_init_wpad<<<(DIM * NPAD + 255) / 256, 256, 0, stream>>>(w_out, wpad);

  k_embed_ln<<<NPOS / 4, 128, 0, stream>>>(tokens, emb, pos, in_g, in_b, xbuf);
  for (int l = 0; l < LAYERS; ++l)
    k_layer<<<NPOS / 4, 128, 0, stream>>>(xbuf, patt, ln_g, ln_b, hp, vp);
  k_final<<<NPOS / 16, 128, 0, stream>>>(xbuf, out_g, out_b, wpad, b_out, out);
}
